// MultiHeadAttention_18966575579699
// MI455X (gfx1250) — matrix operand order repaired and checked by reading, not compiled, not run on hardware
//
#include <hip/hip_runtime.h>
#include <hip/hip_bf16.h>

// MI455X / gfx1250, wave32. All matmuls via v_wmma_f32_16x16x32_bf16.
// Ping-pong (unroll-by-2) software pipelining: fragment double-buffers are
// register-renamed, never copied, so VMEM latency overlaps the WMMA pipe.
//
// Shapes: B=8, S=512, D_MODEL=1024, NHEAD=16, HEAD_DIM=64, M=B*S=4096.

typedef __attribute__((ext_vector_type(16))) __bf16 bf16x16;
typedef __attribute__((ext_vector_type(8)))  __bf16 bf16x8;
typedef __attribute__((ext_vector_type(8)))  float  f32x8;

#define NB 8
#define SEQ 512
#define DM 1024
#define NH 16
#define HD 64
#define MTOT (NB*SEQ)      // 4096
#define N_QKV (3*DM)       // 3072

__device__ __forceinline__ f32x8 zero8() {
  f32x8 z;
#pragma unroll
  for (int i = 0; i < 8; ++i) z[i] = 0.0f;
  return z;
}

__device__ __forceinline__ f32x8 wmma_bf16(bf16x16 a, bf16x16 b, f32x8 c) {
  return __builtin_amdgcn_wmma_f32_16x16x32_bf16(
      /*neg_a=*/false, a, /*neg_b=*/false, b,
      /*c_mod=*/(short)0, c, /*reuse_a=*/false, /*reuse_b=*/false);
}

// A-fragment (16x32, row-major A[M][K], ld in elements, 16-elem aligned).
// Lane L holds row m = L%16; half h = L/16; elems 0..7 -> k = h*8+e,
// elems 8..15 -> k = 16 + h*8 + (e-8).  (ISA 7.12.2, 16-bit A 16x32)
__device__ __forceinline__ bf16x16 load_frag_a(const __bf16* __restrict__ p, int ld) {
  const int lane = threadIdx.x & 31;
  const __bf16* q = p + (size_t)(lane & 15) * ld + ((lane >> 4) << 3);
  bf16x8 lo = *(const bf16x8*)(q);
  bf16x8 hi = *(const bf16x8*)(q + 16);
  bf16x16 r;
#pragma unroll
  for (int i = 0; i < 8; ++i) { r[i] = lo[i]; r[i + 8] = hi[i]; }
  return r;
}

// B-fragment (32x16) from B stored transposed as Bt[N][K] ("NT" gemm).
// Lane L holds col n = L%16; half h = L/16; elems 0..7 -> k = h*8+e,
// elems 8..15 -> k = 16 + h*8 + (e-8).
__device__ __forceinline__ bf16x16 load_frag_b(const __bf16* __restrict__ p, int ld) {
  const int lane = threadIdx.x & 31;
  const __bf16* q = p + (size_t)(lane & 15) * ld + ((lane >> 4) << 3);
  bf16x8 lo = *(const bf16x8*)(q);
  bf16x8 hi = *(const bf16x8*)(q + 16);
  bf16x16 r;
#pragma unroll
  for (int i = 0; i < 8; ++i) { r[i] = lo[i]; r[i + 8] = hi[i]; }
  return r;
}

// -------------------------------------------------- shared 32x64 GEMM core
// acc[2][4] (+=) Arow[32 x DM] @ Brow[64 x DM]^T, ping-pong pipelined.
__device__ __forceinline__ void gemm_core_32x64(
    const __bf16* __restrict__ Arow, const __bf16* __restrict__ Brow,
    f32x8 acc[2][4]) {
  bf16x16 a0[2], b0[4], a1[2], b1[4];
#pragma unroll
  for (int mi = 0; mi < 2; ++mi)
    a0[mi] = load_frag_a(Arow + (size_t)(mi * 16) * DM, DM);
#pragma unroll
  for (int j = 0; j < 4; ++j)
    b0[j] = load_frag_b(Brow + (size_t)(j * 16) * DM, DM);

  for (int k = 0; k < DM; k += 64) {
    // issue buf1 loads (k+32 always in range since k <= DM-64)
#pragma unroll
    for (int mi = 0; mi < 2; ++mi)
      a1[mi] = load_frag_a(Arow + (size_t)(mi * 16) * DM + k + 32, DM);
#pragma unroll
    for (int j = 0; j < 4; ++j)
      b1[j] = load_frag_b(Brow + (size_t)(j * 16) * DM + k + 32, DM);
    // compute on buf0 (loaded 12 loads ago)
#pragma unroll
    for (int mi = 0; mi < 2; ++mi)
#pragma unroll
      for (int j = 0; j < 4; ++j)
        acc[mi][j] = wmma_bf16(a0[mi], b0[j], acc[mi][j]);
    // issue buf0 loads for k+64 (clamped on the last iteration)
    const int kn = (k + 64 < DM) ? (k + 64) : k;
#pragma unroll
    for (int mi = 0; mi < 2; ++mi)
      a0[mi] = load_frag_a(Arow + (size_t)(mi * 16) * DM + kn, DM);
#pragma unroll
    for (int j = 0; j < 4; ++j)
      b0[j] = load_frag_b(Brow + (size_t)(j * 16) * DM + kn, DM);
    // compute on buf1
#pragma unroll
    for (int mi = 0; mi < 2; ++mi)
#pragma unroll
      for (int j = 0; j < 4; ++j)
        acc[mi][j] = wmma_bf16(a1[mi], b1[j], acc[mi][j]);
  }
}

// ---------------------------------------------------------------- converts
__global__ void cvt_f32_to_bf16(const float* __restrict__ x,
                                __bf16* __restrict__ y, int n) {
  int i = blockIdx.x * blockDim.x + threadIdx.x;
  if (i < n) y[i] = (__bf16)x[i];
}

// W[K][N] (row-major) -> Wt[N][K] bf16
__global__ void transpose_to_bf16(const float* __restrict__ W,
                                  __bf16* __restrict__ Wt, int K, int N) {
  int i = blockIdx.x * blockDim.x + threadIdx.x;
  if (i >= K * N) return;
  int k = i / N, n = i - k * N;
  Wt[(size_t)n * K + k] = (__bf16)W[i];
}

// ------------------------------------------------------ QKV GEMM + scatter
// C[4096,3072] = Xb[4096,1024] @ Wqkv_t[3072,1024]^T + bias.
// Wave tile 32(M) x 64(N); 8 waves/WG span 512 N-cols sharing the same
// 32 A-rows. Scatter epilogue: Q/K [bh][s][d], V transposed [bh][d][s].
__global__ void __launch_bounds__(256)
qkv_gemm_kernel(const __bf16* __restrict__ Xb, const __bf16* __restrict__ Wt,
                const float* __restrict__ bias,
                __bf16* __restrict__ Qb, __bf16* __restrict__ Kb,
                __bf16* __restrict__ Vt) {
  const int wave = threadIdx.x >> 5;
  const int lane = threadIdx.x & 31;
  const int m0 = blockIdx.x * 32;
  const int n0 = blockIdx.y * 512 + wave * 64;

  f32x8 acc[2][4];
#pragma unroll
  for (int mi = 0; mi < 2; ++mi)
#pragma unroll
    for (int j = 0; j < 4; ++j) acc[mi][j] = zero8();

  gemm_core_32x64(Xb + (size_t)m0 * DM, Wt + (size_t)n0 * DM, acc);

  // Epilogue: bias + scatter. C layout: VGPR r -> M = r + 8*(lane/16), N = lane%16.
#pragma unroll
  for (int j = 0; j < 4; ++j) {
    const int n = n0 + j * 16 + (lane & 15);
    const int which = n >> 10;          // 0=q 1=k 2=v
    const int f = n & 1023;
    const int h = f >> 6, d = f & 63;
    const float bv = bias[n];
#pragma unroll
    for (int mi = 0; mi < 2; ++mi)
#pragma unroll
      for (int r = 0; r < 8; ++r) {
        const int m = m0 + mi * 16 + r + ((lane >> 4) << 3);
        const int b8 = m >> 9, s = m & 511;
        const size_t bh = (size_t)(b8 * NH + h);
        const float v = acc[mi][j][r] + bv;
        if (which == 0)       Qb[(bh * SEQ + s) * HD + d] = (__bf16)v;
        else if (which == 1)  Kb[(bh * SEQ + s) * HD + d] = (__bf16)v;
        else                  Vt[(bh * HD + d) * SEQ + s] = (__bf16)v;
      }
  }
}

// ------------------------------------------------------------------- RoPE
// In-place on Q,K bf16 [bh][s][64]; pair (d, d+32), d<32.
__global__ void rope_kernel(__bf16* __restrict__ Qb, __bf16* __restrict__ Kb) {
  int idx = blockIdx.x * blockDim.x + threadIdx.x;   // NB*NH*SEQ*32
  if (idx >= NB * NH * SEQ * 32) return;
  const int d  = idx & 31;
  const int s  = (idx >> 5) & (SEQ - 1);
  const int bh = idx >> 14;
  // freq = 10000^(-d/32) = exp(-d * ln(10000)/32)
  const float freq = __expf(-(float)d * (9.210340371976184f / 32.0f));
  const float ang = (float)s * freq;
  float sn, cs;
  __sincosf(ang, &sn, &cs);
  const size_t base = ((size_t)bh * SEQ + s) * HD;
  float q1 = (float)Qb[base + d], q2 = (float)Qb[base + d + 32];
  Qb[base + d]      = (__bf16)(q1 * cs - q2 * sn);
  Qb[base + d + 32] = (__bf16)(q1 * sn + q2 * cs);
  float k1 = (float)Kb[base + d], k2 = (float)Kb[base + d + 32];
  Kb[base + d]      = (__bf16)(k1 * cs - k2 * sn);
  Kb[base + d + 32] = (__bf16)(k1 * sn + k2 * cs);
}

// -------------------------------------------------------------- attention
// Computes S^T = K @ Q^T so the softmaxed P^T C-fragment is *identical* to
// the A-fragment needed for P @ V (A elem e of lane L == st[e>>3][e&7] of
// the same lane). Online softmax, stats per lane (query = lane%16,
// replicated across the two 16-lane halves).

__device__ __forceinline__ void load_kfrags(const __bf16* __restrict__ Kp,
                                            int j0, bf16x16 ka[2][2]) {
#pragma unroll
  for (int t = 0; t < 2; ++t) {
    const __bf16* krow = Kp + (size_t)(j0 + t * 16) * HD;
    ka[t][0] = load_frag_a(krow, HD);
    ka[t][1] = load_frag_a(krow + 32, HD);
  }
}

// One 32-key block: scores via WMMA, online-softmax update, O += P@V.
__device__ __forceinline__ void attn_block(
    const __bf16* __restrict__ Vp, int j0, const bf16x16 ka[2][2],
    const bf16x16 qfb[2], f32x8 o[4], float& m_run, float& l_run, int lane) {
  // V^T B-fragments for this block — issue before the VALU work
  bf16x16 vb[4];
#pragma unroll
  for (int j = 0; j < 4; ++j)
    vb[j] = load_frag_b(Vp + (size_t)(j * 16) * SEQ + j0, SEQ);

  // S^T tiles: t=0 -> keys j0..j0+15, t=1 -> j0+16..j0+31 (M=key, N=query)
  f32x8 st[2];
#pragma unroll
  for (int t = 0; t < 2; ++t) {
    f32x8 z = zero8();
    z = wmma_bf16(ka[t][0], qfb[0], z);
    z = wmma_bf16(ka[t][1], qfb[1], z);
    st[t] = z;
  }

  const float scale = 0.125f;  // 1/sqrt(64)
  float bmax = -1e30f;
#pragma unroll
  for (int t = 0; t < 2; ++t)
#pragma unroll
    for (int r = 0; r < 8; ++r) bmax = fmaxf(bmax, st[t][r]);
  bmax *= scale;
  bmax = fmaxf(bmax, __shfl_xor(bmax, 16, 32));
  const float m_new = fmaxf(m_run, bmax);

  // exponentiate; pack P^T directly as the next A-fragment
  float lsum = 0.0f;
  bf16x16 pa;
#pragma unroll
  for (int t = 0; t < 2; ++t)
#pragma unroll
    for (int r = 0; r < 8; ++r) {
      float p = __expf(st[t][r] * scale - m_new);
      lsum += p;
      pa[t * 8 + r] = (__bf16)p;
    }
  lsum += __shfl_xor(lsum, 16, 32);

  const float corr = __expf(m_run - m_new);
  l_run = l_run * corr + lsum;
  m_run = m_new;

  // rescale O rows: row q' = r + 8*(lane/16); corr(q') lives in lane
  // (lane>=16 ? 24+r : r) of the same half (stats replicated).
  const int srcbase = (lane >> 4) * 24;
  float corr_r[8];
#pragma unroll
  for (int r = 0; r < 8; ++r) corr_r[r] = __shfl(corr, srcbase + r, 32);
#pragma unroll
  for (int j = 0; j < 4; ++j)
#pragma unroll
    for (int r = 0; r < 8; ++r) o[j][r] *= corr_r[r];

  // O += P @ V ;  V^T as B-operand [n=d][k=key]
#pragma unroll
  for (int j = 0; j < 4; ++j) o[j] = wmma_bf16(pa, vb[j], o[j]);
}

__global__ void __launch_bounds__(256)
attn_kernel(const __bf16* __restrict__ Qb, const __bf16* __restrict__ Kb,
            const __bf16* __restrict__ Vt, __bf16* __restrict__ AO) {
  const int wave = threadIdx.x >> 5;
  const int lane = threadIdx.x & 31;
  const int bh = blockIdx.x;                    // 0..127
  const int b = bh >> 4, h = bh & 15;
  const int q0 = blockIdx.y * 128 + wave * 16;  // query tile base

  const __bf16* Qp = Qb + (size_t)bh * SEQ * HD;
  const __bf16* Kp = Kb + (size_t)bh * SEQ * HD;
  const __bf16* Vp = Vt + (size_t)bh * HD * SEQ;

  // Q as B-operand [n=query][k=d], two 32-wide k-chunks; loaded once.
  bf16x16 qfb[2];
#pragma unroll
  for (int kc = 0; kc < 2; ++kc)
    qfb[kc] = load_frag_b(Qp + (size_t)q0 * HD + kc * 32, HD);

  f32x8 o[4];
#pragma unroll
  for (int j = 0; j < 4; ++j) o[j] = zero8();
  float m_run = -1e30f, l_run = 0.0f;

  // ping-pong K fragments across 64-key iterations (no register copies)
  bf16x16 ka0[2][2], ka1[2][2];
  load_kfrags(Kp, 0, ka0);

  for (int j0 = 0; j0 < SEQ; j0 += 64) {
    load_kfrags(Kp, j0 + 32, ka1);                 // always in range
    attn_block(Vp, j0, ka0, qfb, o, m_run, l_run, lane);
    const int jn = (j0 + 64 < SEQ) ? (j0 + 64) : j0;
    load_kfrags(Kp, jn, ka0);                      // clamped on last iter
    attn_block(Vp, j0 + 32, ka1, qfb, o, m_run, l_run, lane);
  }

  // normalize and store AO[m = b*512+q][h*64+d] bf16 (A-form for out proj)
  const float linv = 1.0f / l_run;
  const int srcbase = (lane >> 4) * 24;
  float linv_r[8];
#pragma unroll
  for (int r = 0; r < 8; ++r) linv_r[r] = __shfl(linv, srcbase + r, 32);
#pragma unroll
  for (int j = 0; j < 4; ++j) {
    const int d = j * 16 + (lane & 15);
#pragma unroll
    for (int r = 0; r < 8; ++r) {
      const int q = q0 + r + ((lane >> 4) << 3);
      AO[((size_t)(b * SEQ + q)) * DM + h * HD + d] = (__bf16)(o[j][r] * linv_r[r]);
    }
  }
}

// --------------------------------------------------------- out projection
// out[4096,1024] = AO[4096,1024] @ Wout_t[1024,1024]^T + b_out (f32 out)
__global__ void __launch_bounds__(256)
out_gemm_kernel(const __bf16* __restrict__ AO, const __bf16* __restrict__ Wt,
                const float* __restrict__ bias, float* __restrict__ out) {
  const int wave = threadIdx.x >> 5;
  const int lane = threadIdx.x & 31;
  const int m0 = blockIdx.x * 32;
  const int n0 = blockIdx.y * 512 + wave * 64;

  f32x8 acc[2][4];
#pragma unroll
  for (int mi = 0; mi < 2; ++mi)
#pragma unroll
    for (int j = 0; j < 4; ++j) acc[mi][j] = zero8();

  gemm_core_32x64(AO + (size_t)m0 * DM, Wt + (size_t)n0 * DM, acc);

#pragma unroll
  for (int j = 0; j < 4; ++j) {
    const int n = n0 + j * 16 + (lane & 15);
    const float bv = bias[n];
#pragma unroll
    for (int mi = 0; mi < 2; ++mi)
#pragma unroll
      for (int r = 0; r < 8; ++r) {
        const int m = m0 + mi * 16 + r + ((lane >> 4) << 3);
        out[(size_t)m * DM + n] = acc[mi][j][r] + bv;
      }
  }
}

// ------------------------------------------------------------------ launch
extern "C" void kernel_launch(void* const* d_in, const int* in_sizes, int n_in,
                              void* d_out, int out_size, void* d_ws, size_t ws_size,
                              hipStream_t stream) {
  const float* x    = (const float*)d_in[0];   // [8,512,1024]
  const float* Wqkv = (const float*)d_in[1];   // [1024,3072]
  const float* bqkv = (const float*)d_in[2];   // [3072]
  const float* Wout = (const float*)d_in[3];   // [1024,1024]
  const float* bout = (const float*)d_in[4];   // [1024]
  float* out = (float*)d_out;

  char* ws = (char*)d_ws;
  __bf16* xb   = (__bf16*)(ws);                          //  8 MiB: x bf16 [4096][1024]
  __bf16* wqkt = (__bf16*)(ws + ( 8ull << 20));          //  6 MiB: Wqkv^T [3072][1024]
  __bf16* wot  = (__bf16*)(ws + (14ull << 20));          //  2 MiB: Wout^T [1024][1024]
  __bf16* Qb   = (__bf16*)(ws + (16ull << 20));          //  8 MiB: Q [128][512][64]
  __bf16* Kb   = (__bf16*)(ws + (24ull << 20));          //  8 MiB: K [128][512][64]
  __bf16* Vt   = (__bf16*)(ws + (32ull << 20));          //  8 MiB: V^T [128][64][512]
  __bf16* AO   = (__bf16*)(ws + (40ull << 20));          //  8 MiB: attn out [4096][1024]

  // 1) precision conversion / weight transposes (tiny, bandwidth-trivial)
  cvt_f32_to_bf16<<<(MTOT * DM + 255) / 256, 256, 0, stream>>>(x, xb, MTOT * DM);
  transpose_to_bf16<<<(DM * N_QKV + 255) / 256, 256, 0, stream>>>(Wqkv, wqkt, DM, N_QKV);
  transpose_to_bf16<<<(DM * DM + 255) / 256, 256, 0, stream>>>(Wout, wot, DM, DM);

  // 2) QKV projection + bias + scatter (V stored transposed)
  qkv_gemm_kernel<<<dim3(MTOT / 32, N_QKV / 512), 256, 0, stream>>>(
      xb, wqkt, bqkv, Qb, Kb, Vt);

  // 3) RoPE on Q, K
  rope_kernel<<<(NB * NH * SEQ * 32) / 256, 256, 0, stream>>>(Qb, Kb);

  // 4) attention (flash-style, wmma for both S^T and P·V)
  attn_kernel<<<dim3(NB * NH, SEQ / 128), 256, 0, stream>>>(Qb, Kb, Vt, AO);

  // 5) output projection + bias
  out_gemm_kernel<<<dim3(MTOT / 32, DM / 512), 256, 0, stream>>>(AO, wot, bout, out);
}
